// LlamaAttention_9242769621999
// MI455X (gfx1250) — matrix operand order repaired and checked by reading, not compiled, not run on hardware
//
#include <hip/hip_runtime.h>
#include <hip/hip_bf16.h>

// ---------------------------------------------------------------------------
// Types for CDNA5 WMMA (wave32): v16bf A/B fragments, v8f C/D accumulators.
// ---------------------------------------------------------------------------
typedef __attribute__((ext_vector_type(16))) __bf16 v16bf;
typedef __attribute__((ext_vector_type(8)))  __bf16 v8bf;
typedef __attribute__((ext_vector_type(4)))  __bf16 v4bf;
typedef __attribute__((ext_vector_type(8)))  float  v8f;
typedef __attribute__((ext_vector_type(4)))  int    v4i_t;

__device__ __forceinline__ v16bf pack16(v8bf lo, v8bf hi) {
  v16bf r;
#pragma unroll
  for (int e = 0; e < 8; ++e) { r[e] = lo[e]; r[e + 8] = hi[e]; }
  return r;
}

__device__ __forceinline__ v4bf cvt4(float4 f) {
  v4bf r;
  r[0] = (__bf16)f.x; r[1] = (__bf16)f.y; r[2] = (__bf16)f.z; r[3] = (__bf16)f.w;
  return r;
}

__device__ __forceinline__ v8f wmma_bf16(v16bf a, v16bf b, v8f c) {
  return __builtin_amdgcn_wmma_f32_16x16x32_bf16(
      /*neg_a=*/false, a, /*neg_b=*/false, b,
      /*c_mod=*/(short)0, c, /*reuse_a=*/false, /*reuse_b=*/false);
}

// ---------------------------------------------------------------------------
// Async global->LDS 16B copy (CDNA5 GLOBAL_LOAD_ASYNC_TO_LDS_B128, ASYNCcnt).
// Builtin signature (from clang diagnostic): param1 = addrspace(1) v4i*.
// Guarded: falls back to b128 load + ds_store_b128 if builtins are absent.
// ---------------------------------------------------------------------------
#if defined(__gfx1250__) && __has_builtin(__builtin_amdgcn_global_load_async_to_lds_b128) && __has_builtin(__builtin_amdgcn_s_wait_asynccnt)
#define HAVE_ASYNC_LDS 1
#else
#define HAVE_ASYNC_LDS 0
#endif

typedef __attribute__((address_space(1))) v4i_t* g128ptr;   // global 128b vec
typedef __attribute__((address_space(3))) v4i_t* l128ptr;   // LDS    128b vec

__device__ __forceinline__ void copy16_g2l(const void* g, void* l) {
#if HAVE_ASYNC_LDS
  __builtin_amdgcn_global_load_async_to_lds_b128((g128ptr)g, (l128ptr)l, 0, 0);
#else
  *(v8bf*)l = *(const v8bf*)g;
#endif
}

__device__ __forceinline__ void copy_wait() {
#if HAVE_ASYNC_LDS
  __builtin_amdgcn_s_wait_asynccnt(0);
#endif
}

// ---------------------------------------------------------------------------
// GEMM #1:  Cbf16[M,N] (optionally transposed) = A_f32[M,K] * B_f32[K,N]
// 128x128 tile / workgroup, 256 threads = 8 waves, each wave 32x64.
// TRANS_OUT=true stores C^T at Cb[col*M + row]  (used for V so flash reads V^T)
// ---------------------------------------------------------------------------
#define BM 128
#define BN 128
#define BK 32
#define LDP (BK + 8)   // padded LDS row stride: 40 bf16 = 80B (16B aligned)

template <bool TRANS_OUT>
__global__ __launch_bounds__(256) void gemm_f32_bf16_kernel(
    const float* __restrict__ A, const float* __restrict__ B,
    __bf16* __restrict__ Cb, int M, int N, int K)
{
  __shared__ __align__(16) __bf16 As[BM][LDP];
  __shared__ __align__(16) __bf16 Bt[BN][LDP];   // B tile, transposed (N x K)

  const int tid  = threadIdx.x;
  const int lane = tid & 31;
  const int wid  = tid >> 5;
  const int wm   = wid & 3;
  const int wn   = wid >> 2;
  const int l16  = lane & 15;
  const int hs   = lane >> 4;
  const int row0 = blockIdx.x * BM;
  const int col0 = blockIdx.y * BN;

  v8f acc[2][4];
#pragma unroll
  for (int tm = 0; tm < 2; ++tm)
#pragma unroll
    for (int tn = 0; tn < 4; ++tn)
      acc[tm][tn] = (v8f){0.f,0.f,0.f,0.f,0.f,0.f,0.f,0.f};

  for (int k0 = 0; k0 < K; k0 += BK) {
    __syncthreads();
    float4 abuf[4], bbuf[4];
#pragma unroll
    for (int i = 0; i < 4; ++i) {            // A tile: 128x32 = 1024 float4
      int idx = i * 256 + tid;
      int r = idx >> 3, c4 = idx & 7;
      abuf[i] = *(const float4*)&A[(size_t)(row0 + r) * K + k0 + c4 * 4];
    }
#pragma unroll
    for (int i = 0; i < 4; ++i) {            // B tile: 32x128 = 1024 float4
      int idx = i * 256 + tid;
      int kr = idx >> 5, c4 = idx & 31;
      bbuf[i] = *(const float4*)&B[(size_t)(k0 + kr) * N + col0 + c4 * 4];
    }
    if (k0 + BK < K) {
      __builtin_prefetch(&A[(size_t)(row0 + (tid >> 1)) * K + k0 + BK], 0, 1);
      __builtin_prefetch(&B[(size_t)(k0 + BK + (tid >> 7)) * N + col0 + (tid & 127)], 0, 1);
    }
#pragma unroll
    for (int i = 0; i < 4; ++i) {
      int idx = i * 256 + tid;
      int r = idx >> 3, c4 = idx & 7;
      *(v4bf*)&As[r][c4 * 4] = cvt4(abuf[i]);
    }
#pragma unroll
    for (int i = 0; i < 4; ++i) {            // transposed scatter (LDS-side)
      int idx = i * 256 + tid;
      int kr = idx >> 5, c4 = idx & 31;
      v4bf cv = cvt4(bbuf[i]);
      Bt[c4 * 4 + 0][kr] = cv[0];
      Bt[c4 * 4 + 1][kr] = cv[1];
      Bt[c4 * 4 + 2][kr] = cv[2];
      Bt[c4 * 4 + 3][kr] = cv[3];
    }
    __syncthreads();

    v16bf afrag[2];
#pragma unroll
    for (int tm = 0; tm < 2; ++tm) {
      int r  = wm * 32 + tm * 16 + l16;
      int kb = hs * 8;
      v8bf lo = *reinterpret_cast<const v8bf*>(&As[r][kb]);
      v8bf hi = *reinterpret_cast<const v8bf*>(&As[r][kb + 16]);
      afrag[tm] = pack16(lo, hi);
    }
    v16bf bfrag[4];
#pragma unroll
    for (int tn = 0; tn < 4; ++tn) {
      int n  = wn * 64 + tn * 16 + l16;
      int kb = hs * 8;
      v8bf lo = *reinterpret_cast<const v8bf*>(&Bt[n][kb]);
      v8bf hi = *reinterpret_cast<const v8bf*>(&Bt[n][kb + 16]);
      bfrag[tn] = pack16(lo, hi);
    }
#pragma unroll
    for (int tm = 0; tm < 2; ++tm)
#pragma unroll
      for (int tn = 0; tn < 4; ++tn)
        acc[tm][tn] = wmma_bf16(afrag[tm], bfrag[tn], acc[tm][tn]);
  }

#pragma unroll
  for (int tm = 0; tm < 2; ++tm)
#pragma unroll
    for (int tn = 0; tn < 4; ++tn)
#pragma unroll
      for (int r = 0; r < 8; ++r) {
        int row = row0 + wm * 32 + tm * 16 + r + hs * 8;
        int col = col0 + wn * 64 + tn * 16 + l16;
        if (TRANS_OUT) Cb[(size_t)col * M + row] = (__bf16)acc[tm][tn][r];
        else           Cb[(size_t)row * N + col] = (__bf16)acc[tm][tn][r];
      }
}

// ---------------------------------------------------------------------------
// GEMM #2:  C_f32[M,N] = A_bf16[M,K] * B_f32[K,N]   (output projection)
// A tile is a pure bf16 copy -> async global->LDS path.
// ---------------------------------------------------------------------------
__global__ __launch_bounds__(256) void gemm_bf16A_f32_kernel(
    const __bf16* __restrict__ A, const float* __restrict__ B,
    float* __restrict__ C, int M, int N, int K)
{
  __shared__ __align__(16) __bf16 As[BM][LDP];
  __shared__ __align__(16) __bf16 Bt[BN][LDP];

  const int tid  = threadIdx.x;
  const int lane = tid & 31;
  const int wid  = tid >> 5;
  const int wm   = wid & 3;
  const int wn   = wid >> 2;
  const int l16  = lane & 15;
  const int hs   = lane >> 4;
  const int row0 = blockIdx.x * BM;
  const int col0 = blockIdx.y * BN;

  v8f acc[2][4];
#pragma unroll
  for (int tm = 0; tm < 2; ++tm)
#pragma unroll
    for (int tn = 0; tn < 4; ++tn)
      acc[tm][tn] = (v8f){0.f,0.f,0.f,0.f,0.f,0.f,0.f,0.f};

  for (int k0 = 0; k0 < K; k0 += BK) {
    __syncthreads();
    // A tile: 128x32 bf16 = 512 16B-chunks; async copy, no conversion
#pragma unroll
    for (int i = 0; i < 2; ++i) {
      int idx = i * 256 + tid;
      int r = idx >> 2, ch = idx & 3;       // 4 chunks of 8 bf16 per row
      copy16_g2l(&A[(size_t)(row0 + r) * K + k0 + ch * 8], &As[r][ch * 8]);
    }
    // B tile (f32 weights): batched loads then cvt + transposed scatter
    float4 bbuf[4];
#pragma unroll
    for (int i = 0; i < 4; ++i) {
      int idx = i * 256 + tid;
      int kr = idx >> 5, c4 = idx & 31;
      bbuf[i] = *(const float4*)&B[(size_t)(k0 + kr) * N + col0 + c4 * 4];
    }
    if (k0 + BK < K) {
      __builtin_prefetch(&B[(size_t)(k0 + BK + (tid >> 7)) * N + col0 + (tid & 127)], 0, 1);
    }
#pragma unroll
    for (int i = 0; i < 4; ++i) {
      int idx = i * 256 + tid;
      int kr = idx >> 5, c4 = idx & 31;
      v4bf cv = cvt4(bbuf[i]);
      Bt[c4 * 4 + 0][kr] = cv[0];
      Bt[c4 * 4 + 1][kr] = cv[1];
      Bt[c4 * 4 + 2][kr] = cv[2];
      Bt[c4 * 4 + 3][kr] = cv[3];
    }
    copy_wait();
    __syncthreads();

    v16bf afrag[2];
#pragma unroll
    for (int tm = 0; tm < 2; ++tm) {
      int r  = wm * 32 + tm * 16 + l16;
      int kb = hs * 8;
      v8bf lo = *reinterpret_cast<const v8bf*>(&As[r][kb]);
      v8bf hi = *reinterpret_cast<const v8bf*>(&As[r][kb + 16]);
      afrag[tm] = pack16(lo, hi);
    }
    v16bf bfrag[4];
#pragma unroll
    for (int tn = 0; tn < 4; ++tn) {
      int n  = wn * 64 + tn * 16 + l16;
      int kb = hs * 8;
      v8bf lo = *reinterpret_cast<const v8bf*>(&Bt[n][kb]);
      v8bf hi = *reinterpret_cast<const v8bf*>(&Bt[n][kb + 16]);
      bfrag[tn] = pack16(lo, hi);
    }
#pragma unroll
    for (int tm = 0; tm < 2; ++tm)
#pragma unroll
      for (int tn = 0; tn < 4; ++tn)
        acc[tm][tn] = wmma_bf16(afrag[tm], bfrag[tn], acc[tm][tn]);
  }

#pragma unroll
  for (int tm = 0; tm < 2; ++tm)
#pragma unroll
    for (int tn = 0; tn < 4; ++tn)
#pragma unroll
      for (int r = 0; r < 8; ++r) {
        int row = row0 + wm * 32 + tm * 16 + r + hs * 8;
        int col = col0 + wn * 64 + tn * 16 + l16;
        C[(size_t)row * N + col] = acc[tm][tn][r];
      }
}

// ---------------------------------------------------------------------------
// RoPE on bf16 buffer (in place), with output scale (softmax scale for Q,
// since the scalar commutes with the rotation; 1.0 for K).
// ---------------------------------------------------------------------------
__global__ void rope_bf16_kernel(__bf16* __restrict__ buf, const int* __restrict__ pos_ids,
                                 int nheads, int total, int rowstride, float outscale)
{
  int t = blockIdx.x * blockDim.x + threadIdx.x;
  if (t >= total) return;
  int i = t & 63;
  int rem = t >> 6;
  int head = rem % nheads;
  int token = rem / nheads;
  float pos = (float)pos_ids[token];
  float f = pos * __expf(-(float)i * (1.0f / 64.0f) * 9.210340371976184f);
  float c = __cosf(f), sn = __sinf(f);
  size_t base = (size_t)token * rowstride + head * 128 + i;
  float x1 = (float)buf[base], x2 = (float)buf[base + 64];
  buf[base]      = (__bf16)((x1 * c - x2 * sn) * outscale);
  buf[base + 64] = (__bf16)((x2 * c + x1 * sn) * outscale);
}

// ---------------------------------------------------------------------------
// Flash attention (causal, GQA 32q/8kv heads, head_dim=128), all-bf16 inputs.
// Q: [tokens, 4096] bf16 (scale pre-folded). K: [tokens, 1024] bf16.
// V: TRANSPOSED global [1024, tokens] bf16 -> V tiles copy straight to LDS.
// Output: bf16 token-major [tokens, 4096] feeding the bf16-A O-projection.
// Grid: (s/64, b*32). 128 threads = 4 waves; each wave owns 16 query rows.
// ---------------------------------------------------------------------------
#define FS 2048
#define FH 4096
#define FKV 1024
#define HD 128

__global__ __launch_bounds__(128) void flash_attn_kernel(
    const __bf16* __restrict__ Q, const __bf16* __restrict__ K,
    const __bf16* __restrict__ Vt, __bf16* __restrict__ O)
{
  __shared__ __align__(16) __bf16 Ks[32][HD + 8];      // keys x dims
  __shared__ __align__(16) __bf16 Vl[HD][32 + 8];      // dims x keys (V^T tile)
  __shared__ __align__(16) __bf16 Pt[4][16][32 + 8];   // per-wave P tile

  const int tid  = threadIdx.x;
  const int lane = tid & 31;
  const int wave = tid >> 5;
  const int l16  = lane & 15;
  const int hs   = lane >> 4;

  const int bh  = blockIdx.y;
  const int b   = bh >> 5;
  const int h   = bh & 31;
  const int kvh = h >> 2;
  const int q0  = blockIdx.x * 64;
  const int qw  = q0 + wave * 16;

  // Q fragments (A layout): direct bf16 vector loads, zero converts
  v16bf qf[4];
  {
    const __bf16* qp = Q + (size_t)(b * FS + qw + l16) * FH + h * HD;
    const int kb = hs * 8;
#pragma unroll
    for (int c = 0; c < 4; ++c) {
      int d0 = c * 32 + kb;
      v8bf lo = *(const v8bf*)&qp[d0];
      v8bf hi = *(const v8bf*)&qp[d0 + 16];
      qf[c] = pack16(lo, hi);
    }
  }

  v8f accO[8];
#pragma unroll
  for (int j = 0; j < 8; ++j) accO[j] = (v8f){0.f,0.f,0.f,0.f,0.f,0.f,0.f,0.f};
  float mrow[8], lrow[8];
#pragma unroll
  for (int r = 0; r < 8; ++r) { mrow[r] = -1e30f; lrow[r] = 0.f; }

  const int nkb = (q0 + 64) / 32;
  for (int kb = 0; kb < nkb; ++kb) {
    const int kstart = kb * 32;
    __syncthreads();
    // K tile: 32x128 bf16 = 512 16B-chunks; pure copy -> async LDS path
#pragma unroll
    for (int i = 0; i < 4; ++i) {
      int idx = i * 128 + tid;
      int r = idx >> 4, ch = idx & 15;       // 16 chunks per key row
      copy16_g2l(&K[(size_t)(b * FS + kstart + r) * FKV + kvh * HD + ch * 8],
                 &Ks[r][ch * 8]);
    }
    // V^T tile: 128 dims x 32 keys = 512 16B-chunks; pure copy (transposed
    // in global already, so rows are contiguous along keys)
#pragma unroll
    for (int i = 0; i < 4; ++i) {
      int idx = i * 128 + tid;
      int d = idx >> 2, ch = idx & 3;        // 4 chunks of 8 keys per dim row
      copy16_g2l(&Vt[(size_t)(kvh * HD + d) * (2 * FS) + b * FS + kstart + ch * 8],
                 &Vl[d][ch * 8]);
    }
    copy_wait();
    __syncthreads();

    // S = Q K^T : 2 n-tiles of 16 keys, 4 K-dim chunks each
    v8f sacc[2];
    sacc[0] = (v8f){0.f,0.f,0.f,0.f,0.f,0.f,0.f,0.f};
    sacc[1] = (v8f){0.f,0.f,0.f,0.f,0.f,0.f,0.f,0.f};
#pragma unroll
    for (int j = 0; j < 2; ++j) {
      int krow = j * 16 + l16;
#pragma unroll
      for (int c = 0; c < 4; ++c) {
        int d0 = c * 32 + hs * 8;
        v8bf lo = *reinterpret_cast<const v8bf*>(&Ks[krow][d0]);
        v8bf hi = *reinterpret_cast<const v8bf*>(&Ks[krow][d0 + 16]);
        sacc[j] = wmma_bf16(qf[c], pack16(lo, hi), sacc[j]);
      }
    }

    // Causal mask + online softmax (row = 16 lanes of a half-wave)
#pragma unroll
    for (int r = 0; r < 8; ++r) {
      int qpos = qw + r + hs * 8;
      float s0 = (kstart + l16      <= qpos) ? sacc[0][r] : -1e30f;
      float s1 = (kstart + 16 + l16 <= qpos) ? sacc[1][r] : -1e30f;
      float mx = fmaxf(s0, s1);
#pragma unroll
      for (int off = 1; off < 16; off <<= 1) mx = fmaxf(mx, __shfl_xor(mx, off, 32));
      float mnew = fmaxf(mrow[r], mx);
      float e0 = __expf(s0 - mnew);
      float e1 = __expf(s1 - mnew);
      float corr = __expf(mrow[r] - mnew);
      float rs = e0 + e1;
#pragma unroll
      for (int off = 1; off < 16; off <<= 1) rs += __shfl_xor(rs, off, 32);
      lrow[r] = lrow[r] * corr + rs;
      mrow[r] = mnew;
#pragma unroll
      for (int j = 0; j < 8; ++j) accO[j][r] *= corr;
      int m = r + hs * 8;
      Pt[wave][m][l16]      = (__bf16)e0;
      Pt[wave][m][16 + l16] = (__bf16)e1;
    }

    // P (A layout) via per-wave LDS round-trip; same-wave LDS is in-order
    v16bf pf;
    {
      int kbb = hs * 8;
      v8bf lo = *reinterpret_cast<const v8bf*>(&Pt[wave][l16][kbb]);
      v8bf hi = *reinterpret_cast<const v8bf*>(&Pt[wave][l16][kbb + 16]);
      pf = pack16(lo, hi);
    }
    // O += P V : 8 dim-tiles; contiguous B fragments from V^T tile
#pragma unroll
    for (int j = 0; j < 8; ++j) {
      int n  = j * 16 + l16;
      int e0i = hs * 8;
      v8bf lo = *reinterpret_cast<const v8bf*>(&Vl[n][e0i]);
      v8bf hi = *reinterpret_cast<const v8bf*>(&Vl[n][e0i + 16]);
      accO[j] = wmma_bf16(pf, pack16(lo, hi), accO[j]);
    }
  }

  // Normalize; store bf16 token-major so the bf16-A O-proj GEMM consumes it
#pragma unroll
  for (int r = 0; r < 8; ++r) {
    float inv = 1.0f / lrow[r];
    size_t base = (size_t)(b * FS + qw + r + hs * 8) * FH + h * HD;
#pragma unroll
    for (int j = 0; j < 8; ++j)
      O[base + j * 16 + l16] = (__bf16)(accO[j][r] * inv);
  }
}

// ---------------------------------------------------------------------------
// Launch: QKV proj (bf16 out, V transposed) -> RoPE -> flash -> O proj.
// ---------------------------------------------------------------------------
extern "C" void kernel_launch(void* const* d_in, const int* in_sizes, int n_in,
                              void* d_out, int out_size, void* d_ws, size_t ws_size,
                              hipStream_t stream) {
  (void)in_sizes; (void)n_in; (void)out_size; (void)ws_size;
  const float* X   = (const float*)d_in[0];
  const int*   pid = (const int*)d_in[1];
  const float* wq  = (const float*)d_in[2];
  const float* wk  = (const float*)d_in[3];
  const float* wv  = (const float*)d_in[4];
  const float* wo  = (const float*)d_in[5];
  float* out = (float*)d_out;

  constexpr int B = 2, S = 2048, H = 4096, KV = 1024;
  const int M = B * S;  // 4096 tokens

  __bf16* Qb  = (__bf16*)d_ws;                 // [M, 4096]
  __bf16* Kb  = Qb  + (size_t)M * H;           // [M, 1024]
  __bf16* Vtg = Kb  + (size_t)M * KV;          // [1024, M]  (transposed)
  __bf16* Ob  = Vtg + (size_t)M * KV;          // [M, 4096]

  dim3 blk(256);
  gemm_f32_bf16_kernel<false><<<dim3(M / BM, H  / BN), blk, 0, stream>>>(X, wq, Qb, M, H,  H);
  gemm_f32_bf16_kernel<false><<<dim3(M / BM, KV / BN), blk, 0, stream>>>(X, wk, Kb, M, KV, H);
  gemm_f32_bf16_kernel<true ><<<dim3(M / BM, KV / BN), blk, 0, stream>>>(X, wv, Vtg, M, KV, H);

  const float qscale = 0.08838834764831845f;   // 1/sqrt(128), commutes w/ RoPE
  int totQ = M * 32 * 64;
  rope_bf16_kernel<<<(totQ + 255) / 256, 256, 0, stream>>>(Qb, pid, 32, totQ, H,  qscale);
  int totK = M * 8 * 64;
  rope_bf16_kernel<<<(totK + 255) / 256, 256, 0, stream>>>(Kb, pid, 8,  totK, KV, 1.0f);

  flash_attn_kernel<<<dim3(S / 64, B * 32), dim3(128), 0, stream>>>(Qb, Kb, Vtg, Ob);

  gemm_bf16A_f32_kernel<<<dim3(M / BM, H / BN), blk, 0, stream>>>(Ob, wo, out, M, H, H);
}
